// VariationalGCNEncoder_46583215292646
// MI455X (gfx1250) — matrix operand order repaired and checked by reading, not compiled, not run on hardware
//
#include <hip/hip_runtime.h>

// ---------------------------------------------------------------------------
// VariationalGCNEncoder for MI455X (gfx1250, wave32, WMMA)
//  - GEMMs: v_wmma_f32_16x16x32_bf16, operands pre-converted/pre-packed to
//    bf16 so fragments load as b128 (global for A, LDS-staged for B)
//  - Sparse aggregation: global_atomic_add_f32 (L2-resident working set)
// ---------------------------------------------------------------------------

typedef __attribute__((ext_vector_type(16))) __bf16 v16bf;
typedef __attribute__((ext_vector_type(8)))  float  v8f;

#define NN   50000
#define NE   400000
#define INC  512
#define HIDC 512
#define OUTC 256

// round-to-nearest-even f32 -> bf16, packed pair (lo in [15:0], hi in [31:16])
__device__ __forceinline__ unsigned int f2bf_pair(float lo, float hi) {
    unsigned int ulo = __float_as_uint(lo);
    unsigned int uhi = __float_as_uint(hi);
    ulo += 0x7fffu + ((ulo >> 16) & 1u);
    uhi += 0x7fffu + ((uhi >> 16) & 1u);
    return (ulo >> 16) | (uhi & 0xffff0000u);
}

// ---------------------------------------------------------------------------
// f32 -> bf16 row-major bulk convert (8 elems / thread)
// ---------------------------------------------------------------------------
__global__ void gcn_cvt_bf16(const float* __restrict__ in, __bf16* __restrict__ out, int n8) {
    int i = blockIdx.x * blockDim.x + threadIdx.x;
    if (i >= n8) return;
    const float4 a = ((const float4*)in)[2 * (size_t)i];
    const float4 b = ((const float4*)in)[2 * (size_t)i + 1];
    uint4 o;
    o.x = f2bf_pair(a.x, a.y); o.y = f2bf_pair(a.z, a.w);
    o.z = f2bf_pair(b.x, b.y); o.w = f2bf_pair(b.z, b.w);
    ((uint4*)out)[i] = o;
}

// leaky_relu(h) -> bf16 (8 elems / thread)
__global__ void gcn_leaky_cvt(const float* __restrict__ h, __bf16* __restrict__ out, int n8) {
    int i = blockIdx.x * blockDim.x + threadIdx.x;
    if (i >= n8) return;
    float v[8];
    const float4 a = ((const float4*)h)[2 * (size_t)i];
    const float4 b = ((const float4*)h)[2 * (size_t)i + 1];
    v[0]=a.x; v[1]=a.y; v[2]=a.z; v[3]=a.w; v[4]=b.x; v[5]=b.y; v[6]=b.z; v[7]=b.w;
    #pragma unroll
    for (int k = 0; k < 8; ++k) v[k] = v[k] > 0.0f ? v[k] : 0.01f * v[k];
    uint4 o;
    o.x = f2bf_pair(v[0], v[1]); o.y = f2bf_pair(v[2], v[3]);
    o.z = f2bf_pair(v[4], v[5]); o.w = f2bf_pair(v[6], v[7]);
    ((uint4*)out)[i] = o;
}

// ---------------------------------------------------------------------------
// Pack weights W[K][N] f32 -> Bpack bf16 [K/32][N][32] (K-local contiguous).
// One thread per (ktile, n) column: 32 strided reads -> 64B contiguous write.
// ---------------------------------------------------------------------------
template<int K_DIM, int N_DIM>
__global__ void gcn_pack_w(const float* __restrict__ W, __bf16* __restrict__ Bp) {
    int idx = blockIdx.x * blockDim.x + threadIdx.x;
    if (idx >= (K_DIM / 32) * N_DIM) return;
    const int kt = idx / N_DIM;
    const int n  = idx % N_DIM;
    unsigned int o[16];
    #pragma unroll
    for (int p = 0; p < 16; ++p) {
        const float lo = W[(size_t)(kt * 32 + 2 * p)     * N_DIM + n];
        const float hi = W[(size_t)(kt * 32 + 2 * p + 1) * N_DIM + n];
        o[p] = f2bf_pair(lo, hi);
    }
    uint4* dst = (uint4*)(Bp + (size_t)idx * 32);
    dst[0] = make_uint4(o[0],  o[1],  o[2],  o[3]);
    dst[1] = make_uint4(o[4],  o[5],  o[6],  o[7]);
    dst[2] = make_uint4(o[8],  o[9],  o[10], o[11]);
    dst[3] = make_uint4(o[12], o[13], o[14], o[15]);
}

// ---------------------------------------------------------------------------
// GEMM: C[M x N_DIM] = A[M x K_DIM](bf16) * Bpack(bf16 fragment-major)
// Block = 256 threads = 8 waves. Wave tile: 32x64 (8 WMMA accumulators).
// Block tile: 256 rows x 64 cols. B staged through LDS (4KB / K-step).
// ---------------------------------------------------------------------------
template<int K_DIM, int N_DIM>
__global__ __launch_bounds__(256)
void gcn_gemm_wmma(const __bf16* __restrict__ A, const __bf16* __restrict__ Bpack,
                   float* __restrict__ C, int M) {
    __shared__ uint4 ldsB[256];                       // 64 cols x 32 K x bf16 = 4KB

    const int lane    = threadIdx.x & 31;
    const int wave    = threadIdx.x >> 5;
    const int waveRow = blockIdx.x * 256 + wave * 32;
    const int colBase = blockIdx.y * 64;

    const int  lrow   = lane & 15;
    const bool hiHalf = lane >= 16;
    const int  aSub   = hiHalf ? 8  : 0;              // A half-lane K offset
    const int  bSub   = hiHalf ? 8  : 0;              // B half-lane K offset

    int m0 = waveRow + lrow;        int mc0 = (m0 < M) ? m0 : (M - 1);
    int m1 = waveRow + 16 + lrow;   int mc1 = (m1 < M) ? m1 : (M - 1);

    v8f acc0[4] = {v8f{}, v8f{}, v8f{}, v8f{}};
    v8f acc1[4] = {v8f{}, v8f{}, v8f{}, v8f{}};

    const __bf16* ldsb = (const __bf16*)ldsB;

    for (int kt = 0; kt < K_DIM / 32; ++kt) {
        // ---- cooperative stage of B tile (contiguous 4KB) into LDS ----
        const uint4* bsrc = (const uint4*)(Bpack + ((size_t)kt * N_DIM + colBase) * 32);
        ldsB[threadIdx.x] = bsrc[threadIdx.x];
        if (kt + 1 < K_DIM / 32)                       // prefetch next weight tile
            __builtin_prefetch(Bpack + ((size_t)(kt + 1) * N_DIM + colBase) * 32, 0, 3);
        __syncthreads();

        // ---- A fragments: two b128 loads each (bf16 row-major) ----
        union { v16bf v; uint4 q[2]; } a0, a1;
        const __bf16* ar0 = A + (size_t)mc0 * K_DIM + kt * 32 + aSub;
        const __bf16* ar1 = A + (size_t)mc1 * K_DIM + kt * 32 + aSub;
        a0.q[0] = *(const uint4*)(ar0);  a0.q[1] = *(const uint4*)(ar0 + 16);
        a1.q[0] = *(const uint4*)(ar1);  a1.q[1] = *(const uint4*)(ar1 + 16);

        // ---- 4 N-tiles: b128 LDS reads + 2 WMMAs each ----
        #pragma unroll
        for (int t = 0; t < 4; ++t) {
            union { v16bf v; uint4 q[2]; } bf;
            const __bf16* bl = ldsb + ((t * 16 + lrow) * 32 + bSub);
            bf.q[0] = *(const uint4*)(bl);
            bf.q[1] = *(const uint4*)(bl + 16);
            acc0[t] = __builtin_amdgcn_wmma_f32_16x16x32_bf16(
                false, a0.v, false, bf.v, (short)0, acc0[t], false, false);
            acc1[t] = __builtin_amdgcn_wmma_f32_16x16x32_bf16(
                false, a1.v, false, bf.v, (short)0, acc1[t], false, false);
        }
        __syncthreads();
    }

    // ---- store: lane -> N = colBase + t*16 + lrow; VGPR v -> M row ----
    #pragma unroll
    for (int t = 0; t < 4; ++t) {
        const int n = colBase + t * 16 + lrow;
        #pragma unroll
        for (int vv = 0; vv < 8; ++vv) {
            const int r0 = waveRow + vv + (hiHalf ? 8 : 0);
            const int r1 = r0 + 16;
            if (r0 < M) C[(size_t)r0 * N_DIM + n] = acc0[t][vv];
            if (r1 < M) C[(size_t)r1 * N_DIM + n] = acc1[t][vv];
        }
    }
}

// ---------------------------------------------------------------------------
// Degree / normalization helpers
// ---------------------------------------------------------------------------
__global__ void gcn_deg_init(float* __restrict__ deg, int n) {
    int i = blockIdx.x * blockDim.x + threadIdx.x;
    if (i < n) deg[i] = 1.0f;                          // self-loop contribution
}

__global__ void gcn_deg_count(const long long* __restrict__ ei,
                              float* __restrict__ deg, int e) {
    int i = blockIdx.x * blockDim.x + threadIdx.x;
    if (i < e) atomicAdd(&deg[(int)ei[(size_t)NE + i]], 1.0f);
}

__global__ void gcn_deg_rsqrt(float* __restrict__ deg, int n) {
    int i = blockIdx.x * blockDim.x + threadIdx.x;
    if (i < n) deg[i] = rsqrtf(deg[i]);                // deg >= 1 always
}

// out = bias + dinv^2 * t   (bias + self-loop term)
template<int F>
__global__ void gcn_init_out(const float* __restrict__ t, const float* __restrict__ dinv,
                             const float* __restrict__ bias, float* __restrict__ out, int n) {
    int idx = blockIdx.x * blockDim.x + threadIdx.x;
    if (idx >= n * (F / 4)) return;
    int node = idx / (F / 4);
    int f4   = (idx % (F / 4)) * 4;
    float w  = dinv[node] * dinv[node];
    float4 v = *(const float4*)(t + (size_t)node * F + f4);
    float4 b = *(const float4*)(bias + f4);
    float4 o;
    o.x = b.x + v.x * w; o.y = b.y + v.y * w;
    o.z = b.z + v.z * w; o.w = b.w + v.w * w;
    *(float4*)(out + (size_t)node * F + f4) = o;
}

// out[dst] += (dinv[src]*dinv[dst]) * t[src]  — per (edge, 4-feature group)
template<int F>
__global__ void gcn_scatter(const long long* __restrict__ ei, const float* __restrict__ t,
                            const float* __restrict__ dinv, float* __restrict__ out, int e) {
    int edge = blockIdx.x * blockDim.x + threadIdx.x;
    if (edge >= e) return;
    int g = blockIdx.y * 4;
    int s = (int)ei[edge];
    int d = (int)ei[(size_t)NE + edge];
    float w = dinv[s] * dinv[d];
    float4 v = *(const float4*)(t + (size_t)s * F + g);
    float* o = out + (size_t)d * F + g;
    atomicAdd(o + 0, v.x * w);
    atomicAdd(o + 1, v.y * w);
    atomicAdd(o + 2, v.z * w);
    atomicAdd(o + 3, v.w * w);
}

// ---------------------------------------------------------------------------
extern "C" void kernel_launch(void* const* d_in, const int* in_sizes, int n_in,
                              void* d_out, int out_size, void* d_ws, size_t ws_size,
                              hipStream_t stream) {
    const float*     x   = (const float*)d_in[0];
    const long long* ei  = (const long long*)d_in[1];
    const float*     W1  = (const float*)d_in[2];
    const float*     b1  = (const float*)d_in[3];
    const float*     Wmu = (const float*)d_in[4];
    const float*     bmu = (const float*)d_in[5];
    const float*     Wls = (const float*)d_in[6];
    const float*     bls = (const float*)d_in[7];

    // ---- workspace layout ----
    float* ws   = (float*)d_ws;
    float* dinv = ws;                                   // 50000 f32 (pad 50176)
    float* t1   = ws + 50176;                           // [NN,HIDC] f32
    float* h    = t1 + (size_t)NN * HIDC;               // [NN,HIDC] f32
    __bf16* xbf  = (__bf16*)(h + (size_t)NN * HIDC);    // [NN,INC]  bf16
    __bf16* hbf  = xbf + (size_t)NN * INC;              // [NN,HIDC] bf16
    __bf16* w1p  = hbf + (size_t)NN * HIDC;             // (512/32)*512*32 bf16
    __bf16* wmup = w1p + (INC / 32) * HIDC * 32;
    __bf16* wlsp = wmup + (HIDC / 32) * OUTC * 32;
    float* t2   = t1;                                   // reuse t1 region
    float* t3   = t1 + (size_t)NN * OUTC;

    float* mu = (float*)d_out;                          // [NN,OUTC]
    float* ls = mu + (size_t)NN * OUTC;

    const int B = 256;
    const int gemmGridX = (NN + 255) / 256;             // 196

    // 1) deg -> dinv
    gcn_deg_init <<<(NN + B - 1) / B, B, 0, stream>>>(dinv, NN);
    gcn_deg_count<<<(NE + B - 1) / B, B, 0, stream>>>(ei, dinv, NE);
    gcn_deg_rsqrt<<<(NN + B - 1) / B, B, 0, stream>>>(dinv, NN);

    // 2) one-time precision conversion / weight packing
    gcn_cvt_bf16<<<((NN * INC / 8) + B - 1) / B, B, 0, stream>>>(x, xbf, NN * INC / 8);
    gcn_pack_w<INC,  HIDC><<<(((INC / 32) * HIDC) + B - 1) / B, B, 0, stream>>>(W1,  w1p);
    gcn_pack_w<HIDC, OUTC><<<(((HIDC / 32) * OUTC) + B - 1) / B, B, 0, stream>>>(Wmu, wmup);
    gcn_pack_w<HIDC, OUTC><<<(((HIDC / 32) * OUTC) + B - 1) / B, B, 0, stream>>>(Wls, wlsp);

    // 3) t1 = x @ W1 (WMMA)
    gcn_gemm_wmma<INC, HIDC><<<dim3(gemmGridX, HIDC / 64), B, 0, stream>>>(xbf, w1p, t1, NN);

    // 4) h = b1 + dinv^2*t1 ; edge scatter ; leaky_relu -> bf16
    gcn_init_out<HIDC><<<((NN * (HIDC / 4)) + B - 1) / B, B, 0, stream>>>(t1, dinv, b1, h, NN);
    gcn_scatter <HIDC><<<dim3((NE + B - 1) / B, HIDC / 4), B, 0, stream>>>(ei, t1, dinv, h, NE);
    gcn_leaky_cvt<<<((NN * HIDC / 8) + B - 1) / B, B, 0, stream>>>(h, hbf, NN * HIDC / 8);

    // 5) heads: t2 = h @ Wmu, t3 = h @ Wls (WMMA)
    gcn_gemm_wmma<HIDC, OUTC><<<dim3(gemmGridX, OUTC / 64), B, 0, stream>>>(hbf, wmup, t2, NN);
    gcn_gemm_wmma<HIDC, OUTC><<<dim3(gemmGridX, OUTC / 64), B, 0, stream>>>(hbf, wlsp, t3, NN);

    // 6) outputs: bias + self-loop, then edge scatter
    gcn_init_out<OUTC><<<((NN * (OUTC / 4)) + B - 1) / B, B, 0, stream>>>(t2, dinv, bmu, mu, NN);
    gcn_init_out<OUTC><<<((NN * (OUTC / 4)) + B - 1) / B, B, 0, stream>>>(t3, dinv, bls, ls, NN);
    gcn_scatter <OUTC><<<dim3((NE + B - 1) / B, OUTC / 4), B, 0, stream>>>(ei, t2, dinv, mu, NE);
    gcn_scatter <OUTC><<<dim3((NE + B - 1) / B, OUTC / 4), B, 0, stream>>>(ei, t3, dinv, ls, NE);
}
